// Qwen3Attention_18571438588620
// MI455X (gfx1250) — matrix operand order repaired and checked by reading, not compiled, not run on hardware
//
#include <hip/hip_runtime.h>
#include <hip/hip_bf16.h>

// ---------------------------------------------------------------------------
// Qwen3-style attention forward for MI455X (gfx1250).
// bf16 WMMA + f32 softmax; async global->LDS staging for GEMM B tiles.
// ---------------------------------------------------------------------------

typedef __attribute__((ext_vector_type(16))) __bf16 v16bf;
typedef __attribute__((ext_vector_type(8)))  __bf16 v8bf;
typedef __attribute__((ext_vector_type(8)))  float  v8f;
typedef __attribute__((ext_vector_type(4)))  int    v4i;

typedef __attribute__((address_space(1))) v4i g_v4i;   // global 16B payload
typedef __attribute__((address_space(3))) v4i l_v4i;   // LDS 16B payload

#define S_LEN   2048
#define HID     2560
#define NHEADS  16
#define NKVH    4
#define HDIM    128
#define QKV_N   3072          // NH*HD + 2*NKV*HD
#define QDIM    2048          // NH*HD
#define KVDIM   512           // NKV*HD
#define SCALE_Q 0.08838834764831845f   // 128^-0.5

// ---- gfx1250 async global->LDS copy (ASYNCcnt) ----------------------------
#if __has_builtin(__builtin_amdgcn_global_load_async_to_lds_b128) && \
    __has_builtin(__builtin_amdgcn_s_wait_asynccnt)
#define HAVE_ASYNC_LDS 1
#else
#define HAVE_ASYNC_LDS 0
#endif

__device__ __forceinline__ void async_copy16(const __bf16* g, __bf16* l) {
#if HAVE_ASYNC_LDS
  __builtin_amdgcn_global_load_async_to_lds_b128((g_v4i*)g, (l_v4i*)l, 0, 0);
#else
  *reinterpret_cast<v8bf*>(l) = *reinterpret_cast<const v8bf*>(g);
#endif
}
__device__ __forceinline__ void async_wait0() {
#if HAVE_ASYNC_LDS
  __builtin_amdgcn_s_wait_asynccnt(0);
#endif
}

// ---- fragment loaders -----------------------------------------------------
// A (16x32, MxK, 16-bit): lane&15 = M row; kb = 8*(lane>=16);
// elements 0..7 = K kb..kb+7, elements 8..15 = K kb+16..kb+23.
__device__ __forceinline__ v16bf load_fragA(const __bf16* base, int stride, int lane) {
  int r  = lane & 15;
  int kb = (lane >> 4) << 3;
  const __bf16* p = base + (size_t)r * stride + kb;
  union { v16bf v; v8bf h[2]; } u;
  u.h[0] = *reinterpret_cast<const v8bf*>(p);
  u.h[1] = *reinterpret_cast<const v8bf*>(p + 16);
  return u.v;
}

// B (32x16, KxN, 16-bit): lane&15 = N column; kb = 8*(lane>=16);
// elements 0..7 = K kb..kb+7, elements 8..15 = K kb+16..kb+23. `base` points at the N-major
// (B^T row-major) matrix, stride = K-stride of one N row.
__device__ __forceinline__ v16bf load_fragB(const __bf16* base, int stride, int lane) {
  int n  = lane & 15;
  int kb = (lane >> 4) << 3;
  const __bf16* p = base + (size_t)n * stride + kb;
  union { v16bf v; v8bf h[2]; } u;
  u.h[0] = *reinterpret_cast<const v8bf*>(p);
  u.h[1] = *reinterpret_cast<const v8bf*>(p + 16);
  return u.v;
}

__device__ __forceinline__ v8f wmma_bf16(v16bf a, v16bf b, v8f c) {
  return __builtin_amdgcn_wmma_f32_16x16x32_bf16(false, a, false, b, (short)0, c,
                                                 false, false);
}

__device__ __forceinline__ __bf16 f2bf(float f) { return (__bf16)f; }

// ---- f32 -> bf16 conversion ----------------------------------------------
__global__ __launch_bounds__(256) void cvt_bf16_kernel(const float* __restrict__ in,
                                                       __bf16* __restrict__ out, int n) {
  int i = blockIdx.x * 256 + threadIdx.x;
  if (i < n) out[i] = f2bf(in[i]);
}

// ---- generic GEMM: C[MxN] (f32) = A[MxK](bf16) * W[NxK](bf16)^T -----------
// block = 256 threads (8 waves); block tile 128(M) x 128(N); wave = 16 rows.
// B tile (128x32 bf16 = 8KB) staged through double-buffered LDS with async
// copies and shared by all 8 waves; A fragment prefetched in registers.
__global__ __launch_bounds__(256) void gemm_bf16_kernel(
    const __bf16* __restrict__ A, const __bf16* __restrict__ W,
    float* __restrict__ C, int M, int N, int K) {
  __shared__ __align__(16) __bf16 Bsm[2][128][32];

  int tid  = threadIdx.x;
  int lane = tid & 31;
  int wave = tid >> 5;
  int m0 = blockIdx.y * 128 + wave * 16;
  int n0 = blockIdx.x * 128;

  // stage one 128x32 B tile: 512 chunks of 16B; each thread moves 2 chunks
  auto stageB = [&](int k, int buf) {
#pragma unroll
    for (int i = 0; i < 2; ++i) {
      int c   = tid + i * 256;
      int row = c >> 2;        // 0..127 (N within tile)
      int sub = c & 3;         // 16B chunk within 64B row
      async_copy16(W + (size_t)(n0 + row) * K + k + sub * 8,
                   &Bsm[buf][row][sub * 8]);
    }
  };

  stageB(0, 0);
  v16bf a = load_fragA(A + (size_t)m0 * K, K, lane);
  async_wait0();
  __syncthreads();

  v8f acc[8];
#pragma unroll
  for (int t = 0; t < 8; ++t)
#pragma unroll
    for (int r = 0; r < 8; ++r) acc[t][r] = 0.0f;

  int nk = K >> 5;
  for (int i = 0; i < nk; ++i) {
    int cur = i & 1;
    if (i + 1 < nk) stageB((i + 1) * 32, cur ^ 1);   // async fill other buffer
    v16bf a_next;
    if (i + 1 < nk)
      a_next = load_fragA(A + (size_t)m0 * K + (i + 1) * 32, K, lane);

#pragma unroll
    for (int t = 0; t < 8; ++t) {
      v16bf b = load_fragB(&Bsm[cur][t * 16][0], 32, lane);
      acc[t] = wmma_bf16(a, b, acc[t]);
    }
    if (i + 1 < nk) a = a_next;

    async_wait0();      // our async stores to the other buffer done
    __syncthreads();    // everyone done reading cur + staging next
  }

  int col  = lane & 15;
  int half = lane >> 4;
#pragma unroll
  for (int t = 0; t < 8; ++t)
#pragma unroll
    for (int r = 0; r < 8; ++r)
      C[(size_t)(m0 + r + 8 * half) * N + (n0 + t * 16 + col)] = acc[t][r];
}

// ---- fused RMSNorm + RoPE for q and k; writes bf16 (q pre-scaled) ---------
// grid: (S, 20) ; 32 threads. head 0..15 = q heads, 16..19 = k heads.
__global__ __launch_bounds__(32) void qk_norm_rope_kernel(
    const float* __restrict__ qkv, const float* __restrict__ qw,
    const float* __restrict__ kw, const float* __restrict__ cosb,
    const float* __restrict__ sinb, const int* __restrict__ pos,
    __bf16* __restrict__ qout, __bf16* __restrict__ kout) {
  int t = blockIdx.x;
  int h = blockIdx.y;
  int lane = threadIdx.x;
  bool isq = (h < NHEADS);
  const float* src = qkv + (size_t)t * QKV_N + (isq ? h * HDIM : QDIM + (h - NHEADS) * HDIM);
  const float* w = isq ? qw : kw;

  float x0 = src[4 * lane + 0], x1 = src[4 * lane + 1];
  float x2 = src[4 * lane + 2], x3 = src[4 * lane + 3];
  float ss = x0 * x0 + x1 * x1 + x2 * x2 + x3 * x3;
#pragma unroll
  for (int m = 16; m >= 1; m >>= 1) ss += __shfl_xor(ss, m, 32);
  float inv = rsqrtf(ss * (1.0f / HDIM) + 1e-6f);

  float n0 = x0 * inv * w[4 * lane + 0];
  float n1 = x1 * inv * w[4 * lane + 1];
  float n2 = x2 * inv * w[4 * lane + 2];
  float n3 = x3 * inv * w[4 * lane + 3];

  int p = pos[t];
  float c0 = cosb[(size_t)p * 64 + 2 * lane],     s0 = sinb[(size_t)p * 64 + 2 * lane];
  float c1 = cosb[(size_t)p * 64 + 2 * lane + 1], s1 = sinb[(size_t)p * 64 + 2 * lane + 1];

  float o0 = n0 * c0 - n1 * s0, o1 = n0 * s0 + n1 * c0;
  float o2 = n2 * c1 - n3 * s1, o3 = n2 * s1 + n3 * c1;

  if (isq) {
    __bf16* dst = qout + (size_t)t * QDIM + h * HDIM + 4 * lane;
    dst[0] = f2bf(o0 * SCALE_Q); dst[1] = f2bf(o1 * SCALE_Q);
    dst[2] = f2bf(o2 * SCALE_Q); dst[3] = f2bf(o3 * SCALE_Q);
  } else {
    __bf16* dst = kout + (size_t)t * KVDIM + (h - NHEADS) * HDIM + 4 * lane;
    dst[0] = f2bf(o0); dst[1] = f2bf(o1);
    dst[2] = f2bf(o2); dst[3] = f2bf(o3);
  }
}

// ---- V: convert to bf16, store transposed vt[kvh*HD + d][t] ---------------
__global__ __launch_bounds__(256) void v_transpose_kernel(const float* __restrict__ qkv,
                                                          __bf16* __restrict__ vt) {
  int i = blockIdx.x * 256 + threadIdx.x;        // over S_LEN*KVDIM
  if (i < S_LEN * KVDIM) {
    int t = i >> 9;            // token
    int e = i & 511;           // kvh*128 + d
    vt[(size_t)e * S_LEN + t] = f2bf(qkv[(size_t)t * QKV_N + QDIM + KVDIM + e]);
  }
}

// ---- causal flash attention ----------------------------------------------
// grid: (S/64, NH); block = 128 threads = 4 waves; wave owns 16 q rows.
__global__ __launch_bounds__(128) void flash_attn_kernel(
    const __bf16* __restrict__ qbuf,   // [S][2048] (pre-scaled)
    const __bf16* __restrict__ kbuf,   // [S][512]
    const __bf16* __restrict__ vt,     // [512][S]
    __bf16* __restrict__ obuf) {       // [S][2048]
  int lane = threadIdx.x & 31;
  int wave = threadIdx.x >> 5;
  int h    = blockIdx.y;
  int kvh  = h >> 2;
  int qb   = blockIdx.x * 64 + wave * 16;
  int col  = lane & 15;
  int half = lane >> 4;

  __shared__ __align__(16) __bf16 plds[4][16][32];

  // Q fragments for K-dim 0..127 (4 chunks of 32), reused over all key tiles.
  v16bf qf[4];
#pragma unroll
  for (int c = 0; c < 4; ++c)
    qf[c] = load_fragA(qbuf + (size_t)qb * QDIM + h * HDIM + 32 * c, QDIM, lane);

  v8f o[8];
#pragma unroll
  for (int t = 0; t < 8; ++t)
#pragma unroll
    for (int r = 0; r < 8; ++r) o[t][r] = 0.0f;
  float mst[8], lst[8];
#pragma unroll
  for (int r = 0; r < 8; ++r) { mst[r] = -1e30f; lst[r] = 0.0f; }

  int kend = qb + 16;   // causal: keys < qb+16
  for (int kb2 = 0; kb2 < kend; kb2 += 32) {
    // S = Q * K^T for 32 keys (two 16-key column tiles).
    // Batch the 4 fragment loads before each WMMA group so loads pipeline.
    v8f s0, s1;
#pragma unroll
    for (int r = 0; r < 8; ++r) { s0[r] = 0.0f; s1[r] = 0.0f; }
    {
      v16bf bk[4];
#pragma unroll
      for (int c = 0; c < 4; ++c)
        bk[c] = load_fragB(kbuf + (size_t)kb2 * KVDIM + kvh * HDIM + 32 * c, KVDIM, lane);
#pragma unroll
      for (int c = 0; c < 4; ++c) s0 = wmma_bf16(qf[c], bk[c], s0);
#pragma unroll
      for (int c = 0; c < 4; ++c)
        bk[c] = load_fragB(kbuf + (size_t)(kb2 + 16) * KVDIM + kvh * HDIM + 32 * c, KVDIM, lane);
#pragma unroll
      for (int c = 0; c < 4; ++c) s1 = wmma_bf16(qf[c], bk[c], s1);
    }

    // causal mask + online softmax (per-lane rows r+8*half, reduce over 16 lanes)
    float alpha[8];
#pragma unroll
    for (int r = 0; r < 8; ++r) {
      int m = qb + r + 8 * half;
      if (kb2 + col > m)      s0[r] = -1e30f;
      if (kb2 + 16 + col > m) s1[r] = -1e30f;
      float mx = fmaxf(s0[r], s1[r]);
#pragma unroll
      for (int d = 1; d < 16; d <<= 1) mx = fmaxf(mx, __shfl_xor(mx, d, 32));
      float mnew = fmaxf(mst[r], mx);
      alpha[r] = __expf(mst[r] - mnew);
      mst[r] = mnew;
      float p0 = __expf(s0[r] - mnew);
      float p1 = __expf(s1[r] - mnew);
      s0[r] = p0; s1[r] = p1;
      float ps = p0 + p1;
#pragma unroll
      for (int d = 1; d < 16; d <<= 1) ps += __shfl_xor(ps, d, 32);
      lst[r] = lst[r] * alpha[r] + ps;
    }
#pragma unroll
    for (int t = 0; t < 8; ++t)
#pragma unroll
      for (int r = 0; r < 8; ++r) o[t][r] *= alpha[r];

    // P: C-layout -> A-layout through per-wave LDS (same-wave DS ops are in-order)
#pragma unroll
    for (int r = 0; r < 8; ++r) {
      int m = r + 8 * half;
      plds[wave][m][col]      = f2bf(s0[r]);
      plds[wave][m][16 + col] = f2bf(s1[r]);
    }
    v16bf pf = load_fragA(&plds[wave][0][0], 32, lane);

    // O += P * V (B from transposed V: rows = d, contiguous over keys).
    // Process in two groups of 4 so loads batch ahead of the WMMAs.
#pragma unroll
    for (int g = 0; g < 2; ++g) {
      v16bf vf[4];
#pragma unroll
      for (int j = 0; j < 4; ++j)
        vf[j] = load_fragB(vt + (size_t)(kvh * HDIM + 16 * (4 * g + j)) * S_LEN + kb2,
                           S_LEN, lane);
#pragma unroll
      for (int j = 0; j < 4; ++j)
        o[4 * g + j] = wmma_bf16(pf, vf[j], o[4 * g + j]);
    }
  }

  // epilogue: O /= l, store bf16
  float invl[8];
#pragma unroll
  for (int r = 0; r < 8; ++r) invl[r] = 1.0f / lst[r];
#pragma unroll
  for (int t = 0; t < 8; ++t)
#pragma unroll
    for (int r = 0; r < 8; ++r)
      obuf[(size_t)(qb + r + 8 * half) * QDIM + h * HDIM + 16 * t + col] =
          f2bf(o[t][r] * invl[r]);
}

// ---------------------------------------------------------------------------
extern "C" void kernel_launch(void* const* d_in, const int* in_sizes, int n_in,
                              void* d_out, int out_size, void* d_ws, size_t ws_size,
                              hipStream_t stream) {
  const float* hs   = (const float*)d_in[0];
  const float* wqkv = (const float*)d_in[1];
  const float* wo   = (const float*)d_in[2];
  const float* qnw  = (const float*)d_in[3];
  const float* knw  = (const float*)d_in[4];
  const float* cosb = (const float*)d_in[5];
  const float* sinb = (const float*)d_in[6];
  const int*   pos  = (const int*)d_in[9];
  float* out = (float*)d_out;

  char* ws = (char*)d_ws;
  __bf16* h_bf    = (__bf16*)ws; ws += (size_t)S_LEN * HID * 2;
  __bf16* wqkv_bf = (__bf16*)ws; ws += (size_t)QKV_N * HID * 2;
  __bf16* wo_bf   = (__bf16*)ws; ws += (size_t)HID * QDIM * 2;
  float*  qkv     = (float*)ws;  ws += (size_t)S_LEN * QKV_N * 4;
  __bf16* q_bf    = (__bf16*)ws; ws += (size_t)S_LEN * QDIM * 2;
  __bf16* k_bf    = (__bf16*)ws; ws += (size_t)S_LEN * KVDIM * 2;
  __bf16* vt_bf   = (__bf16*)ws; ws += (size_t)KVDIM * S_LEN * 2;
  __bf16* attn_bf = (__bf16*)ws;

  { int n = S_LEN * HID;
    cvt_bf16_kernel<<<(n + 255) / 256, 256, 0, stream>>>(hs, h_bf, n); }
  { int n = QKV_N * HID;
    cvt_bf16_kernel<<<(n + 255) / 256, 256, 0, stream>>>(wqkv, wqkv_bf, n); }
  { int n = HID * QDIM;
    cvt_bf16_kernel<<<(n + 255) / 256, 256, 0, stream>>>(wo, wo_bf, n); }

  gemm_bf16_kernel<<<dim3(QKV_N / 128, S_LEN / 128), 256, 0, stream>>>(
      h_bf, wqkv_bf, qkv, S_LEN, QKV_N, HID);

  qk_norm_rope_kernel<<<dim3(S_LEN, 20), 32, 0, stream>>>(
      qkv, qnw, knw, cosb, sinb, pos, q_bf, k_bf);

  { int n = S_LEN * KVDIM;
    v_transpose_kernel<<<(n + 255) / 256, 256, 0, stream>>>(qkv, vt_bf); }

  flash_attn_kernel<<<dim3(S_LEN / 64, NHEADS), 128, 0, stream>>>(
      q_bf, k_bf, vt_bf, attn_bf);

  gemm_bf16_kernel<<<dim3(HID / 128, S_LEN / 128), 256, 0, stream>>>(
      attn_bf, wo_bf, out, S_LEN, HID, QDIM);
}
